// EncoderModel_33681133535613
// MI455X (gfx1250) — matrix operand order repaired and checked by reading, not compiled, not run on hardware
//
#include <hip/hip_runtime.h>
#include <hip/hip_bf16.h>
#include <math.h>

// Problem constants (match reference)
#define T_STEPS 12
#define BATCH   16
#define NNODE   1024
#define HDIM    64
#define LLAY    2
#define PROJD   256
#define F2H     128   // 2*H

typedef _Float16 v16h __attribute__((ext_vector_type(16)));
typedef _Float16 h8   __attribute__((ext_vector_type(8)));
typedef float    v8f  __attribute__((ext_vector_type(8)));
typedef int      v4i  __attribute__((ext_vector_type(4)));

#if defined(__has_builtin)
#if __has_builtin(__builtin_amdgcn_global_load_async_to_lds_b128)
#define HAVE_ASYNC_LDS 1
#endif
#if __has_builtin(__builtin_amdgcn_s_wait_asynccnt)
#define HAVE_WAIT_ASYNC 1
#endif
#endif

// builtin signature (from hipcc diagnostic): (v4i AS1*, v4i AS3*, Imm int, Imm int)
#define AS1CAST(p) ((__attribute__((address_space(1))) v4i*)(uintptr_t)(p))
#define AS3CAST(p) ((__attribute__((address_space(3))) v4i*)(unsigned)(uintptr_t)(p))

__device__ __forceinline__ void async_wait_all() {
#if defined(HAVE_WAIT_ASYNC)
    __builtin_amdgcn_s_wait_asynccnt(0);
#else
    asm volatile("s_wait_asynccnt 0x0" ::: "memory");
#endif
}

__device__ __forceinline__ float act_apply(float x, int act) {
    switch (act) {
        case 1:  return 1.0f / (1.0f + __expf(-x));   // sigmoid
        case 2:  return tanhf(x);                      // tanh
        case 3:  return fmaxf(x, 0.0f);                // relu
        default: return x;
    }
}

// ---------------------------------------------------------------------------
// Batched GEMM, all-f16 operands:  acc = A[b](MxK, row-major f16)
//                                      x BT[b](NxK, f16, B stored transposed)
// Epilogue: +bias, activation, write f32 C and/or f16 C16 (either may be null).
// Requires M%128==0, N%64==0, K%32==0 (all call sites satisfy this).
// Workgroup: 256 threads, tile 128(M) x 64(N), K-step 32.
// Tiles staged to LDS as contiguous 16B chunks via async global->LDS copies.
// ---------------------------------------------------------------------------
__global__ __launch_bounds__(256)
void gemm_wmma_h(const _Float16* __restrict__ A,  long long sAb,
                 const _Float16* __restrict__ BT, long long sBb,
                 float* __restrict__ C,   long long sCb,
                 _Float16* __restrict__ C16, long long sC16b,
                 int M, int K, int N,
                 const float* __restrict__ bias, int act)
{
    __shared__ __align__(16) _Float16 As[128][40];  // row-major tile (+pad)
    __shared__ __align__(16) _Float16 Bs[64][40];   // col-major tile (+pad)

    const int tid  = threadIdx.x;
    const int lane = tid & 31;
    const int w    = tid >> 5;
    const int gm0  = blockIdx.x * 128;
    const int n0   = blockIdx.y * 64;
    const int b    = blockIdx.z;

    const _Float16* Ab  = A  + (long long)b * sAb;
    const _Float16* BTb = BT + (long long)b * sBb;

    // static chunk assignment (16B = 8 halves per chunk)
    const int arow0 = tid >> 1;                 // A: one row, 2 chunks per thread
    const int ach0  = (tid & 1) * 16;           // halves 0..15 or 16..31
    const int bcol  = tid >> 2;                 // B: one chunk per thread
    const int bch   = (tid & 3) * 8;

    v8f acc[4];
#pragma unroll
    for (int s = 0; s < 4; ++s) acc[s] = {};

    for (int k0 = 0; k0 < K; k0 += 32) {
        // ---- stage tiles ----
        const _Float16* ga0 = Ab  + (long long)(gm0 + arow0) * K + k0 + ach0;
        const _Float16* gb  = BTb + (long long)(n0 + bcol)   * K + k0 + bch;
#if defined(HAVE_ASYNC_LDS)
        __builtin_amdgcn_global_load_async_to_lds_b128(
            AS1CAST(ga0),     AS3CAST(&As[arow0][ach0]),     0, 0);
        __builtin_amdgcn_global_load_async_to_lds_b128(
            AS1CAST(ga0 + 8), AS3CAST(&As[arow0][ach0 + 8]), 0, 0);
        __builtin_amdgcn_global_load_async_to_lds_b128(
            AS1CAST(gb),      AS3CAST(&Bs[bcol][bch]),       0, 0);
        async_wait_all();
#else
        *(h8*)&As[arow0][ach0]     = *(const h8*)ga0;
        *(h8*)&As[arow0][ach0 + 8] = *(const h8*)(ga0 + 8);
        *(h8*)&Bs[bcol][bch]       = *(const h8*)gb;
#endif
        __syncthreads();

        // ---- WMMA fragments (ISA 7.12.2 layouts) ----
        const int hsel = (lane < 16) ? 0 : 1;
        const int lrow = lane & 15;

        v16h a;
        {
            const int row = w * 16 + lrow;
            const int kb  = hsel * 8;   // lanes 0-15: K 0..7/16..23 ; 16-31: 8..15/24..31
            h8 lo = *(const h8*)&As[row][kb];
            h8 hi = *(const h8*)&As[row][16 + kb];
#pragma unroll
            for (int i = 0; i < 8; ++i) { a[i] = lo[i]; a[8 + i] = hi[i]; }
        }
#pragma unroll
        for (int s = 0; s < 4; ++s) {
            const int col = s * 16 + lrow;
            const int kb  = hsel * 8;   // lanes 0-15: K 0..7/16..23 ; 16-31: 8..15/24..31
            h8 lo = *(const h8*)&Bs[col][kb];
            h8 hi = *(const h8*)&Bs[col][16 + kb];
            v16h bf;
#pragma unroll
            for (int i = 0; i < 8; ++i) { bf[i] = lo[i]; bf[8 + i] = hi[i]; }
            acc[s] = __builtin_amdgcn_wmma_f32_16x16x32_f16(
                false, a, false, bf, (short)0, acc[s], false, false);
        }
        __syncthreads();
    }

    // ---- epilogue: C layout (lane 0-15 -> M=v, lane 16-31 -> M=v+8) ----
    float* Cb = C ? (C + (long long)b * sCb) : nullptr;
    _Float16* C16b = C16 ? (C16 + (long long)b * sC16b) : nullptr;
#pragma unroll
    for (int s = 0; s < 4; ++s) {
        const int col = n0 + s * 16 + (lane & 15);
        const float bv = bias ? bias[col] : 0.0f;
#pragma unroll
        for (int v = 0; v < 8; ++v) {
            const int row = gm0 + w * 16 + ((lane < 16) ? v : v + 8);
            const float val = act_apply(acc[s][v] + bv, act);
            const long long off = (long long)row * N + col;
            if (Cb)   Cb[off]   = val;
            if (C16b) C16b[off] = (_Float16)val;
        }
    }
}

// ---------------------------------------------------------------------------
// Input projection: out[row,:] = relu([x,mask] @ Wp1 + bp1) @ Wp2 + bp2
// ---------------------------------------------------------------------------
__global__ __launch_bounds__(256)
void proj_kernel(const float* __restrict__ x, const float* __restrict__ mask,
                 const float* __restrict__ Wp1, const float* __restrict__ bp1,
                 const float* __restrict__ Wp2, const float* __restrict__ bp2,
                 float* __restrict__ proj)
{
    __shared__ float sp[PROJD];
    const long long row = blockIdx.x;
    const int j = threadIdx.x;
    const float x0 = x[row];
    const float x1 = mask[row];
    const float p = x0 * Wp1[j] + x1 * Wp1[PROJD + j] + bp1[j];
    sp[j] = fmaxf(p, 0.0f);
    __syncthreads();
    if (j < HDIM) {
        float acc = bp2[j];
        for (int k = 0; k < PROJD; ++k) acc += sp[k] * Wp2[k * HDIM + j];
        proj[row * HDIM + j] = acc;
    }
}

// f32 -> f16 convert
__global__ void cvt16_kernel(const float* __restrict__ src, _Float16* __restrict__ dst, int n)
{
    const int idx = blockIdx.x * blockDim.x + threadIdx.x;
    if (idx < n) dst[idx] = (_Float16)src[idx];
}

// W[K,N] f32 -> WT[N,K] f16
__global__ void wtrans_kernel(const float* __restrict__ W, _Float16* __restrict__ WT,
                              int K, int N)
{
    const int idx = blockIdx.x * blockDim.x + threadIdx.x;
    if (idx >= K * N) return;
    const int k = idx / N;
    const int n = idx - k * N;
    WT[n * K + k] = (_Float16)W[idx];
}

// xhT[b][f][m] = f<64 ? x[b,m,f] : h[b,m,f-64] * (r ? r[b,m,f-64] : 1)   (f16)
__global__ void build_xhT_kernel(const float* __restrict__ x, const float* __restrict__ h,
                                 const float* __restrict__ r, _Float16* __restrict__ xhT, int n)
{
    const int idx = blockIdx.x * blockDim.x + threadIdx.x;
    if (idx >= n) return;
    const int b   = idx >> 17;          // / (128*1024)
    const int rem = idx & 131071;
    const int f   = rem >> 10;
    const int m   = rem & 1023;
    const int src = ((b << 10) + m) * HDIM + (f & 63);
    float v = (f < HDIM) ? x[src] : (r ? h[src] * r[src] : h[src]);
    xhT[idx] = (_Float16)v;
}

// row-major f16 concat: out[row, 0:64] = a[row], out[row, 64:128] = b[row]
__global__ void concat_h_kernel(const float* __restrict__ a, const float* __restrict__ b,
                                _Float16* __restrict__ out, int n)
{
    const int idx = blockIdx.x * blockDim.x + threadIdx.x;
    if (idx >= n) return;
    const int f   = idx & (F2H - 1);
    const int row = idx >> 7;
    const float v = (f < HDIM) ? a[row * HDIM + f] : b[row * HDIM + (f - HDIM)];
    out[idx] = (_Float16)v;
}

// h = z*h + (1-z)*c
__global__ void gru_update_kernel(float* __restrict__ h, const float* __restrict__ z,
                                  const float* __restrict__ c, int n)
{
    const int idx = blockIdx.x * blockDim.x + threadIdx.x;
    if (idx < n) {
        const float zz = z[idx];
        h[idx] = zz * h[idx] + (1.0f - zz) * c[idx];
    }
}

__global__ void zero_kernel(float* __restrict__ p, int n)
{
    const int idx = blockIdx.x * blockDim.x + threadIdx.x;
    if (idx < n) p[idx] = 0.0f;
}

// ---------------------------------------------------------------------------
extern "C" void kernel_launch(void* const* d_in, const int* in_sizes, int n_in,
                              void* d_out, int out_size, void* d_ws, size_t ws_size,
                              hipStream_t stream)
{
    (void)in_sizes; (void)n_in; (void)out_size; (void)ws_size;

    const float* inputs = (const float*)d_in[0];
    const float* mask   = (const float*)d_in[1];
    const float* adj    = (const float*)d_in[2];
    const float* Wp1    = (const float*)d_in[3];
    const float* bp1    = (const float*)d_in[4];
    const float* Wp2    = (const float*)d_in[5];
    const float* bp2    = (const float*)d_in[6];
    const float* Wz[2]  = {(const float*)d_in[7],  (const float*)d_in[13]};
    const float* Wr[2]  = {(const float*)d_in[8],  (const float*)d_in[14]};
    const float* Wc[2]  = {(const float*)d_in[9],  (const float*)d_in[15]};
    const float* bz[2]  = {(const float*)d_in[10], (const float*)d_in[16]};
    const float* br[2]  = {(const float*)d_in[11], (const float*)d_in[17]};
    const float* bc[2]  = {(const float*)d_in[12], (const float*)d_in[18]};
    const float* Wd1    = (const float*)d_in[19];
    const float* bd1    = (const float*)d_in[20];
    const float* Wd2    = (const float*)d_in[21];
    const float* bd2    = (const float*)d_in[22];
    float* out = (float*)d_out;

    const long long BN = (long long)BATCH * NNODE;   // 16384
    const int nBNH = (int)(BN * HDIM);               // 1,048,576
    const int nBNF = (int)(BN * F2H);                // 2,097,152
    const long long sXH = (long long)NNODE * F2H;    // 131072 per-batch stride

    // ---- workspace carve-up ----
    char* wsb = (char*)d_ws;
    auto takeF = [&](long long n) { float* p = (float*)wsb; wsb += n * sizeof(float); return p; };
    auto takeH = [&](long long n) { _Float16* p = (_Float16*)wsb; wsb += n * sizeof(_Float16); return p; };

    float*    proj  = takeF((long long)T_STEPS * BN * HDIM);
    float*    zb    = takeF(BN * HDIM);
    float*    rb    = takeF(BN * HDIM);
    float*    cb    = takeF(BN * HDIM);
    float*    hF    = takeF((long long)LLAY * BN * HDIM);
    float*    hR    = takeF((long long)LLAY * BN * HDIM);
    _Float16* adj16 = takeH((long long)NNODE * NNODE);
    _Float16* xhT16 = takeH(BN * F2H);
    _Float16* agg16 = takeH(BN * F2H);
    _Float16* WzT[2][2], *WrT[2][2], *WcT[2][2];
    for (int d = 0; d < 2; ++d)
        for (int l = 0; l < LLAY; ++l) {
            WzT[d][l] = takeH(F2H * HDIM);
            WrT[d][l] = takeH(F2H * HDIM);
            WcT[d][l] = takeH(F2H * HDIM);
        }
    _Float16* Wd1T  = takeH(F2H * PROJD);
    _Float16* Wd2T  = takeH(PROJD * HDIM);
    _Float16* hcat16 = takeH(BN * F2H);
    _Float16* dtmp16 = takeH(BN * PROJD);

    const dim3 blk(256);

    // ---- one-time prep (re-done every call for determinism) ----
    cvt16_kernel<<<(NNODE * NNODE + 255) / 256, blk, 0, stream>>>(adj, adj16, NNODE * NNODE);
    for (int d = 0; d < 2; ++d)
        for (int l = 0; l < LLAY; ++l) {
            const long long o = (long long)l * F2H * HDIM;
            wtrans_kernel<<<(F2H * HDIM + 255) / 256, blk, 0, stream>>>(Wz[d] + o, WzT[d][l], F2H, HDIM);
            wtrans_kernel<<<(F2H * HDIM + 255) / 256, blk, 0, stream>>>(Wr[d] + o, WrT[d][l], F2H, HDIM);
            wtrans_kernel<<<(F2H * HDIM + 255) / 256, blk, 0, stream>>>(Wc[d] + o, WcT[d][l], F2H, HDIM);
        }
    wtrans_kernel<<<(F2H * PROJD + 255) / 256, blk, 0, stream>>>(Wd1, Wd1T, F2H, PROJD);
    wtrans_kernel<<<(PROJD * HDIM + 255) / 256, blk, 0, stream>>>(Wd2, Wd2T, PROJD, HDIM);

    // 1) input projection for all timesteps (shared by both directions)
    proj_kernel<<<dim3((unsigned)(T_STEPS * BN)), blk, 0, stream>>>(
        inputs, mask, Wp1, bp1, Wp2, bp2, proj);

    // 2) bidirectional stacked graph-GRU scan
    for (int dir = 0; dir < 2; ++dir) {
        float* h = dir ? hR : hF;
        zero_kernel<<<(LLAY * nBNH + 255) / 256, blk, 0, stream>>>(h, LLAY * nBNH);

        for (int ti = 0; ti < T_STEPS; ++ti) {
            const int t = dir ? (T_STEPS - 1 - ti) : ti;
            const float* x = proj + (long long)t * nBNH;

            for (int l = 0; l < LLAY; ++l) {
                float* hl = h + (long long)l * nBNH;

                // xh^T = [x, h]^T (f16)
                build_xhT_kernel<<<(nBNF + 255) / 256, blk, 0, stream>>>(x, hl, nullptr, xhT16, nBNF);
                // agg = adj @ xh  (batched; adj shared via strideA=0) -> f16
                gemm_wmma_h<<<dim3(NNODE / 128, F2H / 64, BATCH), blk, 0, stream>>>(
                    adj16, 0, xhT16, sXH, nullptr, 0, agg16, sXH,
                    NNODE, NNODE, F2H, nullptr, 0);
                // z = sigmoid(agg @ Wz + bz); r = sigmoid(agg @ Wr + br)
                gemm_wmma_h<<<dim3((unsigned)(BN / 128), 1, 1), blk, 0, stream>>>(
                    agg16, 0, WzT[dir][l], 0, zb, 0, nullptr, 0,
                    (int)BN, F2H, HDIM, bz[dir] + l * HDIM, 1);
                gemm_wmma_h<<<dim3((unsigned)(BN / 128), 1, 1), blk, 0, stream>>>(
                    agg16, 0, WrT[dir][l], 0, rb, 0, nullptr, 0,
                    (int)BN, F2H, HDIM, br[dir] + l * HDIM, 1);
                // xh^T = [x, r*h]^T
                build_xhT_kernel<<<(nBNF + 255) / 256, blk, 0, stream>>>(x, hl, rb, xhT16, nBNF);
                gemm_wmma_h<<<dim3(NNODE / 128, F2H / 64, BATCH), blk, 0, stream>>>(
                    adj16, 0, xhT16, sXH, nullptr, 0, agg16, sXH,
                    NNODE, NNODE, F2H, nullptr, 0);
                // c = tanh(agg @ Wc + bc)
                gemm_wmma_h<<<dim3((unsigned)(BN / 128), 1, 1), blk, 0, stream>>>(
                    agg16, 0, WcT[dir][l], 0, cb, 0, nullptr, 0,
                    (int)BN, F2H, HDIM, bc[dir] + l * HDIM, 2);
                // h = z*h + (1-z)*c
                gru_update_kernel<<<(nBNH + 255) / 256, blk, 0, stream>>>(hl, zb, cb, nBNH);

                x = hl;  // next layer consumes this layer's output
            }
        }
    }

    // 3) decoder per layer: out[l] = relu([hF,hR] @ Wd1 + bd1) @ Wd2 + bd2
    for (int l = 0; l < LLAY; ++l) {
        const float* hf = hF + (long long)l * nBNH;
        const float* hr = hR + (long long)l * nBNH;
        concat_h_kernel<<<(nBNF + 255) / 256, blk, 0, stream>>>(hf, hr, hcat16, nBNF);
        gemm_wmma_h<<<dim3((unsigned)(BN / 128), PROJD / 64, 1), blk, 0, stream>>>(
            hcat16, 0, Wd1T, 0, nullptr, 0, dtmp16, 0,
            (int)BN, F2H, PROJD, bd1, 3);
        gemm_wmma_h<<<dim3((unsigned)(BN / 128), HDIM / 64, 1), blk, 0, stream>>>(
            dtmp16, 0, Wd2T, 0, out + (long long)l * nBNH, 0, nullptr, 0,
            (int)BN, PROJD, HDIM, bd2, 0);
    }
}
